// MultiHeadAttention_23081154248710
// MI455X (gfx1250) — matrix operand order repaired and checked by reading, not compiled, not run on hardware
//
#include <hip/hip_runtime.h>

typedef __attribute__((ext_vector_type(16))) __bf16 v16bf;
typedef __attribute__((ext_vector_type(8)))  __bf16 v8bf;
typedef __attribute__((ext_vector_type(8)))  float  v8f;
typedef __attribute__((ext_vector_type(4)))  unsigned v4u;
typedef __attribute__((ext_vector_type(8)))  unsigned v8u;

#define D_MODEL 1024
#define NHEAD   16
#define DKDIM   64
#define BATCH   2
#define SEQ     2048
#define MROWS   (BATCH*SEQ)      // 4096
#define NEGBIG  (-1.0e9f)

// ---------------------------------------------------------------------------
// helpers: 16-lane row reductions (C-fragment rows span one half of the wave)
// ---------------------------------------------------------------------------
__device__ __forceinline__ float rowmax16(float x) {
  x = fmaxf(x, __shfl_xor(x, 1, 32));
  x = fmaxf(x, __shfl_xor(x, 2, 32));
  x = fmaxf(x, __shfl_xor(x, 4, 32));
  x = fmaxf(x, __shfl_xor(x, 8, 32));
  return x;
}
__device__ __forceinline__ float rowsum16(float x) {
  x += __shfl_xor(x, 1, 32);
  x += __shfl_xor(x, 2, 32);
  x += __shfl_xor(x, 4, 32);
  x += __shfl_xor(x, 8, 32);
  return x;
}

// LDS byte address of a __shared__ object: flat address has the shared
// aperture in bits [63:32], LDS offset in bits [31:0] (ISA 10.2).
__device__ __forceinline__ unsigned lds_addr_of(const void* p) {
  return (unsigned)(unsigned long long)p;
}

// ---------------------------------------------------------------------------
// fp32 -> bf16 conversion (grid-stride)
// ---------------------------------------------------------------------------
__global__ __launch_bounds__(256)
void cvt_f32_bf16(const float* __restrict__ in, __bf16* __restrict__ out, int n) {
  int i = blockIdx.x * blockDim.x + threadIdx.x;
  int stride = gridDim.x * blockDim.x;
  for (; i < n; i += stride) out[i] = (__bf16)in[i];
}

// ---------------------------------------------------------------------------
// NT GEMM: C[M,N] = A[M,K](bf16 rowmajor) x W[N,K]^T(bf16 rowmajor) + bias
// mode 0: write bf16 into head-split [B,H,S,DK]    (QKV projections)
// mode 1: write f32 into [M,N]                      (output projection)
// block = 128 threads = 4 waves, each wave owns a 16x64 output tile.
//   - shared 64x32 W tile: TDM tensor_load_to_lds (one descriptor per k-step,
//     issued by wave 0, tracked with TENSORcnt), double buffered
//   - per-wave 16x32 A tile: global_load_async_to_lds (ASYNCcnt), double
//     buffered, wave-private so no extra barriers
// ---------------------------------------------------------------------------
__global__ __launch_bounds__(128)
void gemm_nt_bf16(const __bf16* __restrict__ A,
                  const __bf16* __restrict__ W,
                  const float*  __restrict__ bias,
                  __bf16* __restrict__ outb,
                  float*  __restrict__ outf,
                  int M, int N, int K, int mode) {
  __shared__ __align__(32) __bf16 wtile[2][64 * 32];      // 2 x 4 KB
  __shared__ __align__(32) __bf16 atile[2][4][16 * 32];   // 2 x 4 x 1 KB

  const int tid  = threadIdx.x;
  const int lane = tid & 31;
  const int wid  = tid >> 5;               // 0..3
  const int l16  = lane & 15;
  const int hv   = lane >> 4;              // which half of the wave
  const int m0   = blockIdx.x * 64 + wid * 16;
  const int n0   = blockIdx.y * 64;

  const unsigned ldsW = lds_addr_of(&wtile[0][0]);
  const unsigned ldsA = lds_addr_of(&atile[0][0][0]);

  // --- TDM staging of the shared W tile: 64 rows x 32 cols bf16, row stride K
  auto stageW = [&](int k0, int buf) {
    unsigned long long ga =
        (unsigned long long)(const void*)(W + (size_t)n0 * K + k0);
    v4u g0;
    g0[0] = 1u;                                        // count=1, user mode
    g0[1] = ldsW + (unsigned)buf * (64 * 32 * 2);      // lds_addr
    g0[2] = (unsigned)ga;                              // global_addr[31:0]
    g0[3] = ((unsigned)(ga >> 32) & 0x01FFFFFFu)       // global_addr[56:32]
            | 0x80000000u;                             // type=2 ("image")
    v8u g1;
    g1[0] = 0x00010000u;        // workgroup_mask=0, data_size=1 (2 bytes)
    g1[1] = 32u << 16;          // tensor_dim0 = 32 (== tile: no OOB clip)
    g1[2] = 64u << 16;          // tensor_dim0[hi]=0, tensor_dim1 = 64
    g1[3] = 32u << 16;          // tensor_dim1[hi]=0, tile_dim0 = 32
    g1[4] = 64u;                // tile_dim1 = 64, tile_dim2 = 0
    g1[5] = (unsigned)K;        // tensor_dim0_stride = K elements
    g1[6] = 0u;
    g1[7] = 0u;
    v4u gz = {0u, 0u, 0u, 0u};  // groups 2/3: dims 2..4 unused
    asm volatile("tensor_load_to_lds %0, %1, %2, %3"
                 :: "s"(g0), "s"(g1), "s"(gz), "s"(gz) : "memory");
  };

  // --- async staging of this wave's 16x32 A tile (2 x 512B per wave)
  auto stageA = [&](int k0, int buf) {
#pragma unroll
    for (int j = 0; j < 2; ++j) {
      int e   = (j * 32 + lane) * 8;      // bf16 element offset in 16x32 tile
      int row = e >> 5;
      int col = e & 31;
      const __bf16* src = A + (size_t)(m0 + row) * K + k0 + col;
      unsigned dst = ldsA + (unsigned)(((buf * 4 + wid) * 512 + e) * 2);
      asm volatile("global_load_async_to_lds_b128 %0, %1, off"
                   :: "v"(dst), "v"(src) : "memory");
    }
  };

  v8f acc[4];
#pragma unroll
  for (int t = 0; t < 4; ++t)
    acc[t] = v8f{0.f, 0.f, 0.f, 0.f, 0.f, 0.f, 0.f, 0.f};

  const int nk = K / 32;
  stageA(0, 0);
  if (wid == 0) stageW(0, 0);

  for (int i = 0; i < nk; ++i) {
    const int kc = i * 32;
    const bool more = (i + 1) < nk;

    if (more) {
      stageA(kc + 32, (i + 1) & 1);
      if (wid == 0) stageW(kc + 32, (i + 1) & 1);
    }

    // both DMA flavors retire in order per wave: waiting for <= (in-flight
    // next-tile count) guarantees the current tile has landed in LDS
    if (more) asm volatile("s_wait_asynccnt 0x2" ::: "memory");
    else      asm volatile("s_wait_asynccnt 0x0" ::: "memory");
    if (wid == 0) {
      if (more) __builtin_amdgcn_s_wait_tensorcnt(1);
      else      __builtin_amdgcn_s_wait_tensorcnt(0);
    }
    __syncthreads();

    // A fragment from LDS: lane = row m (l16), one contiguous 32B chunk per lane
    const __bf16* ab = &atile[i & 1][wid][0];
    v16bf af = *(const v16bf*)(ab + l16 * 32 + hv * 16);

    // B fragments batched so dscnt waits can stagger instead of 0-waits
    const __bf16* buf = &wtile[i & 1][0];
    v16bf b0 = *(const v16bf*)(buf + (0 * 16 + l16) * 32 + hv * 16);
    v16bf b1 = *(const v16bf*)(buf + (1 * 16 + l16) * 32 + hv * 16);
    v16bf b2 = *(const v16bf*)(buf + (2 * 16 + l16) * 32 + hv * 16);
    v16bf b3 = *(const v16bf*)(buf + (3 * 16 + l16) * 32 + hv * 16);
    acc[0] = __builtin_amdgcn_wmma_f32_16x16x32_bf16(
        false, af, false, b0, (short)0, acc[0], false, false);
    acc[1] = __builtin_amdgcn_wmma_f32_16x16x32_bf16(
        false, af, false, b1, (short)0, acc[1], false, false);
    acc[2] = __builtin_amdgcn_wmma_f32_16x16x32_bf16(
        false, af, false, b2, (short)0, acc[2], false, false);
    acc[3] = __builtin_amdgcn_wmma_f32_16x16x32_bf16(
        false, af, false, b3, (short)0, acc[3], false, false);

    __syncthreads();   // all waves done reading before buffers are re-staged
  }

  // epilogue: C layout -> lanes 0-15 rows v, lanes 16-31 rows v+8
#pragma unroll
  for (int t = 0; t < 4; ++t) {
    int n = n0 + t * 16 + l16;
    float bv = bias[n];
#pragma unroll
    for (int v = 0; v < 8; ++v) {
      int m = m0 + v + hv * 8;
      float val = acc[t][v] + bv;
      if (mode == 0) {
        int b  = m >> 11;       // m / SEQ
        int s  = m & 2047;      // m % SEQ
        int h  = n >> 6;        // n / DK
        int dk = n & 63;        // n % DK
        outb[((size_t)(b * NHEAD + h) * SEQ + s) * DKDIM + dk] = (__bf16)val;
      } else {
        outf[(size_t)m * N + n] = val;
      }
    }
  }
}

// ---------------------------------------------------------------------------
// flash attention over head-split bf16 Q/K/V [B,H,S,64]
// grid = (S/64, B*H), block = 128 (4 waves x 16 query rows)
// keys processed 32 at a time: 4 WMMA (QK^T) + 4 WMMA (PV) per chunk.
// K fragments double-buffered in registers; V staged transposed through LDS.
// ---------------------------------------------------------------------------
__global__ __launch_bounds__(128)
void flash_attn(const __bf16* __restrict__ Qp,
                const __bf16* __restrict__ Kp,
                const __bf16* __restrict__ Vp,
                const int*    __restrict__ mask,
                __bf16* __restrict__ ctx) {
  __shared__ __align__(32) __bf16 vt[DKDIM * 32];     // V^T chunk [64][32]
  __shared__ __align__(32) __bf16 pl[4 * 16 * 32];    // per-wave P tiles

  const int lane = threadIdx.x & 31;
  const int wid  = threadIdx.x >> 5;
  const int l16  = lane & 15;
  const int hv   = lane >> 4;
  const int bh   = blockIdx.y;          // batch*head
  const int bi   = bh >> 4;
  const int h    = bh & 15;
  const int q0   = blockIdx.x * 64 + wid * 16;   // this wave's query tile

  const __bf16* Qh = Qp + (size_t)bh * SEQ * DKDIM;
  const __bf16* Kh = Kp + (size_t)bh * SEQ * DKDIM;
  const __bf16* Vh = Vp + (size_t)bh * SEQ * DKDIM;

  // loop-invariant Q fragments (d 0..31 and 32..63)
  const __bf16* qrow = Qh + (size_t)(q0 + l16) * DKDIM;
  v16bf a0, a1;
  {
    a0 = *(const v16bf*)(qrow + hv * 16);
    a1 = *(const v16bf*)(qrow + 32 + hv * 16);
  }

  // K fragments for one 32-key chunk: [subtile][d-half]
  auto loadK = [&](int kb, int sf, v16bf* b01) {
    const __bf16* krow = Kh + (size_t)(kb + sf * 16 + l16) * DKDIM + hv * 16;
    b01[0] = *(const v16bf*)(krow);
    b01[1] = *(const v16bf*)(krow + 32);
  };

  v16bf kcur[2][2], knext[2][2];
  loadK(0, 0, kcur[0]);
  loadK(0, 1, kcur[1]);

  v8f acc[4];
#pragma unroll
  for (int t = 0; t < 4; ++t)
    acc[t] = v8f{0.f, 0.f, 0.f, 0.f, 0.f, 0.f, 0.f, 0.f};
  float rm[8], rl[8];
#pragma unroll
  for (int v = 0; v < 8; ++v) { rm[v] = -3.0e38f; rl[v] = 0.f; }

  __bf16* plw = pl + wid * 16 * 32;
  const float scale = 0.125f;   // 1/sqrt(64)

  for (int kb = 0; kb < SEQ; kb += 32) {
    const bool more = (kb + 32) < SEQ;

    // cooperative transpose-stage of V chunk into LDS: vt[d][k]
    __syncthreads();
    {
      int k  = threadIdx.x >> 2;          // 0..31
      int d0 = (threadIdx.x & 3) * 16;
      v16bf vv = *(const v16bf*)(Vh + (size_t)(kb + k) * DKDIM + d0);
#pragma unroll
      for (int j = 0; j < 16; ++j) vt[(d0 + j) * 32 + k] = vv[j];
    }
    __syncthreads();

    // issue next chunk's K fragments now; consumed next iteration
    if (more) {
      loadK(kb + 32, 0, knext[0]);
      loadK(kb + 32, 1, knext[1]);
      __builtin_prefetch(Kh + (size_t)(kb + 64) * DKDIM, 0, 0);
    }

    // ---- scores for two 16-key subtiles ----
    float sc[2][8];
#pragma unroll
    for (int sf = 0; sf < 2; ++sf) {
      v8f c = {};
      c = __builtin_amdgcn_wmma_f32_16x16x32_bf16(
          false, a0, false, kcur[sf][0], (short)0, c, false, false);
      c = __builtin_amdgcn_wmma_f32_16x16x32_bf16(
          false, a1, false, kcur[sf][1], (short)0, c, false, false);
      int col = kb + sf * 16 + l16;
#pragma unroll
      for (int v = 0; v < 8; ++v) {
        int r = q0 + v + hv * 8;
        int mv = mask[(size_t)r * SEQ + col];
        sc[sf][v] = mv ? c[v] * scale : NEGBIG;
      }
    }

    // ---- online softmax + stash P (bf16) into per-wave LDS tile ----
#pragma unroll
    for (int v = 0; v < 8; ++v) {
      float mx   = rowmax16(fmaxf(sc[0][v], sc[1][v]));
      float mnew = fmaxf(rm[v], mx);
      float alpha = __expf(rm[v] - mnew);
      float p0 = __expf(sc[0][v] - mnew);
      float p1 = __expf(sc[1][v] - mnew);
      rl[v] = rl[v] * alpha + rowsum16(p0 + p1);
      rm[v] = mnew;
#pragma unroll
      for (int t = 0; t < 4; ++t) acc[t][v] *= alpha;
      int pr = (v + hv * 8) * 32 + l16;
      plw[pr]      = (__bf16)p0;
      plw[pr + 16] = (__bf16)p1;
    }
    asm volatile("s_wait_dscnt 0" ::: "memory");

    // P back out in A-fragment layout (16 rows x 32 keys)
    v16bf pfrag = *(const v16bf*)(plw + l16 * 32 + hv * 16);

    // ---- PV: B fragments are contiguous 32B rows of V^T in LDS ----
    v16bf bv0 = *(const v16bf*)(&vt[(0 * 16 + l16) * 32 + hv * 16]);
    v16bf bv1 = *(const v16bf*)(&vt[(1 * 16 + l16) * 32 + hv * 16]);
    v16bf bv2 = *(const v16bf*)(&vt[(2 * 16 + l16) * 32 + hv * 16]);
    v16bf bv3 = *(const v16bf*)(&vt[(3 * 16 + l16) * 32 + hv * 16]);
    acc[0] = __builtin_amdgcn_wmma_f32_16x16x32_bf16(
        false, pfrag, false, bv0, (short)0, acc[0], false, false);
    acc[1] = __builtin_amdgcn_wmma_f32_16x16x32_bf16(
        false, pfrag, false, bv1, (short)0, acc[1], false, false);
    acc[2] = __builtin_amdgcn_wmma_f32_16x16x32_bf16(
        false, pfrag, false, bv2, (short)0, acc[2], false, false);
    acc[3] = __builtin_amdgcn_wmma_f32_16x16x32_bf16(
        false, pfrag, false, bv3, (short)0, acc[3], false, false);

    if (more) {
#pragma unroll
      for (int sf = 0; sf < 2; ++sf) {
        kcur[sf][0] = knext[sf][0];
        kcur[sf][1] = knext[sf][1];
      }
    }
  }

  // epilogue: normalize and scatter bf16 into merged context [B,S,D]
#pragma unroll
  for (int v = 0; v < 8; ++v) {
    float inv = 1.0f / rl[v];
    int s = q0 + v + hv * 8;
    __bf16* orow = ctx + ((size_t)(bi * SEQ + s)) * D_MODEL + h * DKDIM;
#pragma unroll
    for (int t = 0; t < 4; ++t)
      orow[t * 16 + l16] = (__bf16)(acc[t][v] * inv);
  }
}

// ---------------------------------------------------------------------------
// host-side orchestration
// ---------------------------------------------------------------------------
extern "C" void kernel_launch(void* const* d_in, const int* in_sizes, int n_in,
                              void* d_out, int out_size, void* d_ws, size_t ws_size,
                              hipStream_t stream) {
  (void)in_sizes; (void)n_in; (void)out_size; (void)ws_size;

  const float* q    = (const float*)d_in[0];
  const float* k    = (const float*)d_in[1];
  const float* v    = (const float*)d_in[2];
  const int*   mask = (const int*)d_in[3];
  const float* wq_w = (const float*)d_in[4];
  const float* wq_b = (const float*)d_in[5];
  const float* wk_w = (const float*)d_in[6];
  const float* wk_b = (const float*)d_in[7];
  const float* wv_w = (const float*)d_in[8];
  const float* wv_b = (const float*)d_in[9];
  const float* wo_w = (const float*)d_in[10];
  const float* wo_b = (const float*)d_in[11];

  char* ws = (char*)d_ws;
  const size_t MB = 1024ull * 1024ull;
  __bf16* qb  = (__bf16*)(ws + 0 * MB);    // [4096,1024] bf16  (8 MB)
  __bf16* kb  = (__bf16*)(ws + 8 * MB);
  __bf16* vb  = (__bf16*)(ws + 16 * MB);
  __bf16* wqb = (__bf16*)(ws + 24 * MB);   // [1024,1024] bf16 (2 MB each)
  __bf16* wkb = (__bf16*)(ws + 26 * MB);
  __bf16* wvb = (__bf16*)(ws + 28 * MB);
  __bf16* wob = (__bf16*)(ws + 30 * MB);
  __bf16* Qp  = (__bf16*)(ws + 32 * MB);   // [B,H,S,64] bf16 (8 MB each)
  __bf16* Kp  = (__bf16*)(ws + 40 * MB);
  __bf16* Vp  = (__bf16*)(ws + 48 * MB);
  __bf16* ctx = (__bf16*)(ws + 56 * MB);   // [B,S,D] bf16 (8 MB)

  const int nbig = MROWS * D_MODEL;        // 4M elements
  const int nw   = D_MODEL * D_MODEL;      // 1M elements

  cvt_f32_bf16<<<2048, 256, 0, stream>>>(q,    qb,  nbig);
  cvt_f32_bf16<<<2048, 256, 0, stream>>>(k,    kb,  nbig);
  cvt_f32_bf16<<<2048, 256, 0, stream>>>(v,    vb,  nbig);
  cvt_f32_bf16<<<1024, 256, 0, stream>>>(wq_w, wqb, nw);
  cvt_f32_bf16<<<1024, 256, 0, stream>>>(wk_w, wkb, nw);
  cvt_f32_bf16<<<1024, 256, 0, stream>>>(wv_w, wvb, nw);
  cvt_f32_bf16<<<1024, 256, 0, stream>>>(wo_w, wob, nw);

  dim3 ggrid(MROWS / 64, D_MODEL / 64);    // (64,16)
  gemm_nt_bf16<<<ggrid, 128, 0, stream>>>(qb, wqb, wq_b, Qp, nullptr,
                                          MROWS, D_MODEL, D_MODEL, 0);
  gemm_nt_bf16<<<ggrid, 128, 0, stream>>>(kb, wkb, wk_b, Kp, nullptr,
                                          MROWS, D_MODEL, D_MODEL, 0);
  gemm_nt_bf16<<<ggrid, 128, 0, stream>>>(vb, wvb, wv_b, Vp, nullptr,
                                          MROWS, D_MODEL, D_MODEL, 0);

  dim3 agrid(SEQ / 64, BATCH * NHEAD);     // (32,32)
  flash_attn<<<agrid, 128, 0, stream>>>(Qp, Kp, Vp, mask, ctx);

  gemm_nt_bf16<<<ggrid, 128, 0, stream>>>(ctx, wob, wo_b, nullptr, (float*)d_out,
                                          MROWS, D_MODEL, D_MODEL, 1);
}
